// Faust_84086869721204
// MI455X (gfx1250) — matrix operand order repaired and checked by reading, not compiled, not run on hardware
//
#include <hip/hip_runtime.h>
#include <hip/hip_bf16.h>

// ---------------------------------------------------------------------------
// SplineConv net for MI455X (gfx1250, wave32, WMMA).
//  * All GEMMs run on v_wmma_f32_16x16x32_f16; GEMM shapes are template
//    parameters so fragments live in statically-indexed VGPRs (no movrel).
//  * Activations + weights pre-converted to fp16; weights stored transposed
//    so both WMMA fragments are contiguous global_load_b128 pairs.
//  * Z (N*125*64 fp16 = 160 MB) is L2-resident (192 MB global L2) for the
//    random per-edge gather; accumulation fp32 throughout.
// ---------------------------------------------------------------------------

typedef __attribute__((ext_vector_type(16))) _Float16 v16h;
typedef __attribute__((ext_vector_type(8)))  _Float16 v8h;
typedef __attribute__((ext_vector_type(2)))  _Float16 v2h;
typedef __attribute__((ext_vector_type(8)))  float    v8f;

#define KSIZE 5
#define PDIM  3
#define KTOT  125   // 5^3
#define SBAS  8     // 2^3

__device__ __forceinline__ v16h cat16(v8h lo, v8h hi) {
    return __builtin_shufflevector(lo, hi, 0,1,2,3,4,5,6,7,8,9,10,11,12,13,14,15);
}

// ---------------------------------------------------------------------------
// fp32 [B][R][C]  ->  fp16 transposed + K-padded  [B][C][Rpad]  (zero fill).
// ---------------------------------------------------------------------------
__global__ __launch_bounds__(256)
void transpose_f16_kernel(const float* __restrict__ in, _Float16* __restrict__ out,
                          int Bb, int R, int C, int Rpad)
{
    int i = blockIdx.x * blockDim.x + threadIdx.x;
    int total = Bb * C * Rpad;
    if (i >= total) return;
    int r = i % Rpad;
    int c = (i / Rpad) % C;
    int b = i / (Rpad * C);
    out[i] = (r < R) ? (_Float16)in[((size_t)b * R + r) * C + c] : (_Float16)0.0f;
}

// fp32 [N][R] -> fp16 zero-padded [N][Rpad]  (row-major, for activations)
__global__ __launch_bounds__(256)
void pad_f16_kernel(const float* __restrict__ in, _Float16* __restrict__ out,
                    int N, int R, int Rpad)
{
    int i = blockIdx.x * blockDim.x + threadIdx.x;
    if (i >= N * Rpad) return;
    int r = i % Rpad;
    int n = i / Rpad;
    out[i] = (r < R) ? (_Float16)in[(size_t)n * R + r] : (_Float16)0.0f;
}

// ---------------------------------------------------------------------------
// Per-edge degree-1 open B-spline basis + flat kernel index.
// ---------------------------------------------------------------------------
__global__ __launch_bounds__(256)
void spline_basis_kernel(const float* __restrict__ pseudo,
                         float* __restrict__ basis, int* __restrict__ kidx, int E)
{
    int e = blockIdx.x * blockDim.x + threadIdx.x;
    if (e >= E) return;
    float frac[PDIM];
    int   ibot[PDIM];
    #pragma unroll
    for (int d = 0; d < PDIM; ++d) {
        float v = pseudo[e * PDIM + d] * (float)(KSIZE - 1);
        float b = floorf(v);
        frac[d] = v - b;
        ibot[d] = (int)b;
    }
    #pragma unroll
    for (int s = 0; s < SBAS; ++s) {
        float bp = 1.0f;
        int   idx = 0, stride = 1;
        #pragma unroll
        for (int d = 0; d < PDIM; ++d) {
            int off = (s >> d) & 1;
            bp *= off ? frac[d] : (1.0f - frac[d]);
            int id = ibot[d] + off;
            id = id < 0 ? 0 : (id > KSIZE - 1 ? KSIZE - 1 : id);
            idx += id * stride;
            stride *= KSIZE;
        }
        basis[e * SBAS + s] = bp;
        kidx [e * SBAS + s] = idx;
    }
}

// ---------------------------------------------------------------------------
// Fragment loaders (fp16 in memory, all 16B-aligned contiguous loads).
//   A (row-major [M][Kd]):  a[i] = A[row][kk + (i/8)*16 + khalf*8 + i%8]
//   Bt (col-major [N][Kd]): b[i] = Bt[col][kk + (i/8)*16 + khalf*8 + i%8]
// ---------------------------------------------------------------------------
__device__ __forceinline__ v16h load_a_frag(const _Float16* __restrict__ A,
                                            int lda, int row, int kk, int khalf)
{
    const v8h* pa = (const v8h*)(A + (size_t)row * lda + kk + (khalf << 3));
    return cat16(pa[0], pa[2]);   // +0 halves and +16 halves
}
__device__ __forceinline__ v16h load_b_frag(const _Float16* __restrict__ Bt,
                                            int ldb, int col, int kk, int khalf)
{
    const v8h* pb = (const v8h*)(Bt + (size_t)col * ldb + kk + (khalf << 3));
    return cat16(pb[0], pb[2]);   // +0 halves and +16 halves
}

// ---------------------------------------------------------------------------
// Z[n][k][co] = (h[n] @ W[k])[co], fp16 out.  grid = (N/16, KTOT), wave/block.
// Compile-time KD/COUT -> fully unrolled, static fragment registers.
// ---------------------------------------------------------------------------
template<int KD, int COUT>
__global__ __launch_bounds__(32)
void node_weight_gemm(const _Float16* __restrict__ H,   // [N][KD] fp16
                      const _Float16* __restrict__ Wt,  // [K][COUT][KD] fp16
                      _Float16* __restrict__ Z)         // [N][KTOT][COUT]
{
    constexpr int NK = KD / 32;
    int lane  = threadIdx.x & 31;
    int khalf = lane >> 4;
    int tm    = blockIdx.x << 4;
    int k     = blockIdx.y;
    int row   = tm + (lane & 15);
    const _Float16* Bt = Wt + (size_t)k * COUT * KD;

    v16h afrag[NK];
    #pragma unroll
    for (int t = 0; t < NK; ++t)
        afrag[t] = load_a_frag(H, KD, row, t << 5, khalf);

    #pragma unroll
    for (int tn = 0; tn < COUT / 16; ++tn) {
        int col = (tn << 4) + (lane & 15);
        v8f c = {};
        #pragma unroll
        for (int t = 0; t < NK; ++t) {
            v16h b = load_b_frag(Bt, KD, col, t << 5, khalf);
            c = __builtin_amdgcn_wmma_f32_16x16x32_f16(false, afrag[t], false, b,
                                                       (short)0, c, false, false);
        }
        #pragma unroll
        for (int v = 0; v < 8; ++v) {
            int r = tm + (khalf << 3) + v;
            Z[((size_t)r * KTOT + k) * COUT + col] = (_Float16)c[v];
        }
    }
}

// ---------------------------------------------------------------------------
// Per-edge gather from fp16 Z + weighted sum + fp32 atomic scatter into agg.
// One wave per edge; each lane owns channel pairs -> one b32 load per tap.
// ---------------------------------------------------------------------------
template<int COUT>
__global__ __launch_bounds__(256)
void edge_scatter(const _Float16* __restrict__ Z,
                  const float* __restrict__ basis,
                  const int* __restrict__ kidx,
                  const int* __restrict__ src,
                  const int* __restrict__ tgt,
                  float* __restrict__ agg,
                  int E)
{
    int wave = (int)((blockIdx.x * blockDim.x + threadIdx.x) >> 5);
    int lane = threadIdx.x & 31;
    if (wave >= E) return;
    int s0 = src[wave];
    int t0 = tgt[wave];
    float bs[SBAS];
    int   ks[SBAS];
    #pragma unroll
    for (int s = 0; s < SBAS; ++s) {
        bs[s] = basis[wave * SBAS + s];
        ks[s] = kidx [wave * SBAS + s];
    }
    const _Float16* Zn = Z + (size_t)s0 * KTOT * COUT;
    constexpr int PAIRS = COUT / 2;
    #pragma unroll
    for (int pidx = lane; pidx < PAIRS; pidx += 32) {
        float acc0 = 0.0f, acc1 = 0.0f;
        #pragma unroll
        for (int s = 0; s < SBAS; ++s) {
            v2h z = *(const v2h*)(Zn + (size_t)ks[s] * COUT + (pidx << 1));
            acc0 += bs[s] * (float)z[0];
            acc1 += bs[s] * (float)z[1];
        }
        float* ap = agg + (size_t)t0 * COUT + (pidx << 1);
        atomicAdd(ap,     acc0);
        atomicAdd(ap + 1, acc1);
    }
}

// ---------------------------------------------------------------------------
// out = act(A @ Bt^T + bias [+ add]).  act: 0 = none, 1 = ELU.
// Writes fp16 (outh) or fp32 (outf).  grid = (M/16, Ncol/16), wave/block.
// ---------------------------------------------------------------------------
template<int KD>
__global__ __launch_bounds__(32)
void gemm_bias_act(const _Float16* __restrict__ A,    // [M][KD]
                   const _Float16* __restrict__ Bt,   // [Ncol][KD]
                   const float* __restrict__ bias,
                   const float* __restrict__ add,     // nullable, [M][Ncol]
                   _Float16* __restrict__ outh,       // nullable
                   float* __restrict__ outf,          // nullable
                   int Ncol, int act)
{
    int lane  = threadIdx.x & 31;
    int khalf = lane >> 4;
    int tm  = blockIdx.x << 4;
    int tn  = blockIdx.y << 4;
    int row = tm + (lane & 15);
    int col = tn + (lane & 15);

    v8f c = {};
    #pragma unroll
    for (int kk = 0; kk < KD; kk += 32) {
        v16h a = load_a_frag(A, KD, row, kk, khalf);
        v16h b = load_b_frag(Bt, KD, col, kk, khalf);
        c = __builtin_amdgcn_wmma_f32_16x16x32_f16(false, a, false, b,
                                                   (short)0, c, false, false);
    }
    #pragma unroll
    for (int v = 0; v < 8; ++v) {
        int r = tm + (khalf << 3) + v;
        float val = c[v] + bias[col];
        if (add)  val += add[(size_t)r * Ncol + col];
        if (act)  val = val > 0.0f ? val : (expf(val) - 1.0f);
        if (outh) outh[(size_t)r * Ncol + col] = (_Float16)val;
        if (outf) outf[(size_t)r * Ncol + col] = val;
    }
}

// ---------------------------------------------------------------------------
// Host orchestration.
// Input order: 0 x | 1 edge_index | 2 edge_attr | 3..8 Ws | 9..14 roots |
//              15..20 biases | 21 lin1_w | 22 lin1_b | 23 lin2_w | 24 lin2_b
// ---------------------------------------------------------------------------
extern "C" void kernel_launch(void* const* d_in, const int* in_sizes, int n_in,
                              void* d_out, int out_size, void* d_ws, size_t ws_size,
                              hipStream_t stream)
{
    const int N = in_sizes[0] / 14;      // 10000
    const int E = in_sizes[1] / 2;       // 80000

    const float* x         = (const float*)d_in[0];
    const int*   src       = (const int*)d_in[1];
    const int*   tgt       = src + E;
    const float* edge_attr = (const float*)d_in[2];
    const float* Ws[6];    for (int i = 0; i < 6; ++i) Ws[i]     = (const float*)d_in[3 + i];
    const float* roots[6]; for (int i = 0; i < 6; ++i) roots[i]  = (const float*)d_in[9 + i];
    const float* biases[6];for (int i = 0; i < 6; ++i) biases[i] = (const float*)d_in[15 + i];
    const float* lin1_w = (const float*)d_in[21];
    const float* lin1_b = (const float*)d_in[22];
    const float* lin2_w = (const float*)d_in[23];
    const float* lin2_b = (const float*)d_in[24];

    // Workspace carve-out (256B aligned).
    char* p = (char*)d_ws;
    auto carve = [&](size_t bytes) -> void* {
        void* r = (void*)p;
        p += (bytes + 255) & ~(size_t)255;
        return r;
    };
    _Float16* Z     = (_Float16*)carve((size_t)N * KTOT * 64 * 2);   // 160 MB
    float*    basis = (float*)   carve((size_t)E * SBAS * 4);
    int*      kidx  = (int*)     carve((size_t)E * SBAS * 4);
    float*    agg   = (float*)   carve((size_t)N * 64 * 4);
    _Float16* xh    = (_Float16*)carve((size_t)N * 32 * 2);          // padded 14->32
    _Float16* hA    = (_Float16*)carve((size_t)N * 64 * 2);
    _Float16* hB    = (_Float16*)carve((size_t)N * 64 * 2);
    _Float16* mlp   = (_Float16*)carve((size_t)N * 256 * 2);
    _Float16* Wt    = (_Float16*)carve((size_t)KTOT * 64 * 64 * 2);  // per-layer reuse
    _Float16* roott = (_Float16*)carve((size_t)64 * 64 * 2);         // per-layer reuse
    _Float16* lin1t = (_Float16*)carve((size_t)256 * 64 * 2);
    _Float16* lin2t = (_Float16*)carve((size_t)64 * 256 * 2);
    (void)ws_size; (void)n_in; (void)out_size;

    const int dims_in [6] = {14, 32, 64, 64, 64, 64};
    const int dims_pad[6] = {32, 32, 64, 64, 64, 64};   // K-dim padded to mult of 32
    const int dims_out[6] = {32, 64, 64, 64, 64, 64};

    // One-time conversions.
    pad_f16_kernel<<<(N * 32 + 255) / 256, 256, 0, stream>>>(x, xh, N, 14, 32);
    transpose_f16_kernel<<<(256 * 64 + 255) / 256, 256, 0, stream>>>(
        lin1_w, lin1t, 1, 64, 256, 64);
    transpose_f16_kernel<<<(64 * 256 + 255) / 256, 256, 0, stream>>>(
        lin2_w, lin2t, 1, 256, 64, 256);
    spline_basis_kernel<<<(E + 255) / 256, 256, 0, stream>>>(edge_attr, basis, kidx, E);

    const _Float16* h = xh;
    _Float16* ping = hA;
    _Float16* pong = hB;
    for (int l = 0; l < 6; ++l) {
        int cin  = dims_in[l];
        int kd   = dims_pad[l];
        int cout = dims_out[l];

        // Weights -> transposed/padded fp16.
        int wtn = KTOT * cout * kd;
        transpose_f16_kernel<<<(wtn + 255) / 256, 256, 0, stream>>>(
            Ws[l], Wt, KTOT, cin, cout, kd);
        int rtn = cout * kd;
        transpose_f16_kernel<<<(rtn + 255) / 256, 256, 0, stream>>>(
            roots[l], roott, 1, cin, cout, kd);

        // Z = h @ W[l]   (WMMA, fp16 output kept hot in L2).
        dim3 zgrid(N / 16, KTOT);
        if (l == 0)      node_weight_gemm<32, 32><<<zgrid, 32, 0, stream>>>(h, Wt, Z);
        else if (l == 1) node_weight_gemm<32, 64><<<zgrid, 32, 0, stream>>>(h, Wt, Z);
        else             node_weight_gemm<64, 64><<<zgrid, 32, 0, stream>>>(h, Wt, Z);

        // agg = 0; scatter edge messages.
        hipMemsetAsync(agg, 0, (size_t)N * cout * 4, stream);
        if (cout == 32)
            edge_scatter<32><<<(E + 7) / 8, 256, 0, stream>>>(
                Z, basis, kidx, src, tgt, agg, E);
        else
            edge_scatter<64><<<(E + 7) / 8, 256, 0, stream>>>(
                Z, basis, kidx, src, tgt, agg, E);

        // h_next = ELU(agg + h @ root + bias)   (WMMA).
        dim3 ggrid(N / 16, cout / 16);
        if (kd == 32)
            gemm_bias_act<32><<<ggrid, 32, 0, stream>>>(
                h, roott, biases[l], agg, ping, nullptr, cout, /*act=*/1);
        else
            gemm_bias_act<64><<<ggrid, 32, 0, stream>>>(
                h, roott, biases[l], agg, ping, nullptr, cout, /*act=*/1);

        h = ping;
        _Float16* t = ping; ping = pong; pong = t;
    }

    // MLP head: ELU(h @ lin1 + b1) @ lin2 + b2  (final output fp32).
    gemm_bias_act<64><<<dim3(N / 16, 256 / 16), 32, 0, stream>>>(
        h, lin1t, lin1_b, nullptr, mlp, nullptr, 256, /*act=*/1);
    gemm_bias_act<256><<<dim3(N / 16, 64 / 16), 32, 0, stream>>>(
        mlp, lin2t, lin2_b, nullptr, nullptr, (float*)d_out, 64, /*act=*/0);
}
